// GATModel_31885837205854
// MI455X (gfx1250) — matrix operand order repaired and checked by reading, not compiled, not run on hardware
//
#include <hip/hip_runtime.h>

// ---------------------------------------------------------------------------
// Types for CDNA5 WMMA (wave32)
// ---------------------------------------------------------------------------
typedef __bf16 bf16_t;
typedef __attribute__((ext_vector_type(16))) __bf16 v16bf;
typedef __attribute__((ext_vector_type(8)))  __bf16 bf16x8;
typedef __attribute__((ext_vector_type(8)))  float  v8f;
typedef __attribute__((ext_vector_type(4)))  int    v4i;

// gfx1250 async global->LDS DMA (ASYNCcnt-tracked), if the toolchain has it
#if __has_builtin(__builtin_amdgcn_global_load_async_to_lds_b128) && \
    __has_builtin(__builtin_amdgcn_s_wait_asynccnt)
#define HAVE_ASYNC_LDS 1
#else
#define HAVE_ASYNC_LDS 0
#endif

#if HAVE_ASYNC_LDS
typedef __attribute__((address_space(1))) v4i glob_v4i;
typedef __attribute__((address_space(3))) v4i lds_v4i;
__device__ __forceinline__ void async_cp16(const bf16_t* g, bf16_t* l) {
    __builtin_amdgcn_global_load_async_to_lds_b128(
        (glob_v4i*)(g), (lds_v4i*)(l), 0, 0);
}
#endif

// ---------------------------------------------------------------------------
// Helpers
// ---------------------------------------------------------------------------
__device__ __forceinline__ unsigned f2ord(float f) {
    unsigned u = __float_as_uint(f);
    return (u & 0x80000000u) ? ~u : (u | 0x80000000u);
}
__device__ __forceinline__ float ord2f(unsigned u) {
    return (u & 0x80000000u) ? __uint_as_float(u & 0x7fffffffu)
                             : __uint_as_float(~u);
}
__device__ __forceinline__ float lrelu02(float v) { return v >= 0.f ? v : 0.2f * v; }
__device__ __forceinline__ float selu(float v) {
    const float sc = 1.0507009873554805f, al = 1.6732632423543772f;
    return v > 0.f ? sc * v : sc * al * (__expf(v) - 1.f);
}

// ---------------------------------------------------------------------------
// fp32 -> bf16 convert with K padding (rows x K -> rows x Kpad, zero-padded)
// ---------------------------------------------------------------------------
__global__ void k_f32_to_bf16_pad(const float* __restrict__ s, bf16_t* __restrict__ d,
                                  long long rows, int K, int Kpad) {
    long long total = rows * (long long)Kpad;
    for (long long i = blockIdx.x * (long long)blockDim.x + threadIdx.x; i < total;
         i += (long long)gridDim.x * blockDim.x) {
        long long r = i / Kpad;
        int c = (int)(i - r * Kpad);
        d[i] = (c < K) ? (bf16_t)s[r * (long long)K + c] : (bf16_t)0.0f;
    }
}

__global__ void k_fill_u32(unsigned* __restrict__ p, unsigned v, long long n) {
    for (long long i = blockIdx.x * (long long)blockDim.x + threadIdx.x; i < n;
         i += (long long)gridDim.x * blockDim.x)
        p[i] = v;
}

// ---------------------------------------------------------------------------
// bf16 WMMA GEMM:  C[M,Nout] = A[M,Kpad] @ Bw[Nout,Kpad]^T   (fp32 accumulate)
// 256 threads = 8 waves; block tile 64x64; each wave a 16x32 tile (2 WMMAs).
// Double-buffered LDS; async global->LDS DMA when available.
// ---------------------------------------------------------------------------
__global__ __launch_bounds__(256) void k_gemm_bf16(
    const bf16_t* __restrict__ A, const bf16_t* __restrict__ Bw,
    float* __restrict__ C, int M, int Nout, int Kpad, int nTilesN) {
    __shared__ __attribute__((aligned(16))) bf16_t lA[2][64 * 32];  // 2 x 4 KB
    __shared__ __attribute__((aligned(16))) bf16_t lB[2][64 * 32];  // 2 x 4 KB

    const int tid  = threadIdx.x;
    const int wave = tid >> 5, lane = tid & 31;
    const int bm = blockIdx.x / nTilesN, bn = blockIdx.x % nTilesN;
    const int rowBase = bm * 64, colBase = bn * 64;
    const int wr = wave >> 1, wc = wave & 1;     // wave tile: rows wr*16, cols wc*32
    const int r  = lane & 15, kh = lane >> 4;

    // uniform cooperative loaders: thread -> (row/col 0..63, 8-elem K group)
    const int arow = tid >> 2;
    const int kg   = (tid & 3) << 3;
    int gr = rowBase + arow; if (gr >= M)    gr = M - 1;     // clamp, EXEC stays full
    int gc = colBase + arow; if (gc >= Nout) gc = Nout - 1;
    const bf16_t* gA = A  + (size_t)gr * Kpad + kg;
    const bf16_t* gB = Bw + (size_t)gc * Kpad + kg;
    const int sOff = arow * 32 + kg;

    v8f acc0 = {}, acc1 = {};

#if HAVE_ASYNC_LDS
    // prime buffer 0 via async DMA (no VGPR round-trip)
    async_cp16(gA, &lA[0][sOff]);
    async_cp16(gB, &lB[0][sOff]);
    for (int k0 = 0; k0 < Kpad; k0 += 32) {
        const int cur = (k0 >> 5) & 1;
        __builtin_amdgcn_s_wait_asynccnt(0);   // tile k resident in LDS
        __syncthreads();
        if (k0 + 32 < Kpad) {                  // overlap tile k+1 DMA with compute
            async_cp16(gA + k0 + 32, &lA[cur ^ 1][sOff]);
            async_cp16(gB + k0 + 32, &lB[cur ^ 1][sOff]);
            __builtin_prefetch(gA + k0 + 64, 0, 3);
        }
#else
    bf16x8 rA = *(const bf16x8*)gA;
    bf16x8 rB = *(const bf16x8*)gB;
    for (int k0 = 0; k0 < Kpad; k0 += 32) {
        const int cur = (k0 >> 5) & 1;
        *(bf16x8*)(&lA[cur][sOff]) = rA;
        *(bf16x8*)(&lB[cur][sOff]) = rB;
        __syncthreads();
        if (k0 + 32 < Kpad) {                  // stage tile k+1 into registers
            rA = *(const bf16x8*)(gA + k0 + 32);
            rB = *(const bf16x8*)(gB + k0 + 32);
            __builtin_prefetch(gA + k0 + 64, 0, 3);
        }
#endif
        // A fragment: lane=row; elems 0..15 -> K = kh*16 + j (contiguous)
        union { bf16x8 h[2]; v16bf v; } fa, fb0, fb1;
        const bf16_t* ap = &lA[cur][(wr * 16 + r) * 32];
        fa.h[0] = *(const bf16x8*)(ap + kh * 16);
        fa.h[1] = *(const bf16x8*)(ap + kh * 16 + 8);
        // B fragment: lane=col; elems 0..15 -> K = kh*16 + j (contiguous)
        const bf16_t* bp0 = &lB[cur][(wc * 32 + r) * 32];
        const bf16_t* bp1 = bp0 + 16 * 32;
        fb0.h[0] = *(const bf16x8*)(bp0 + kh * 16);
        fb0.h[1] = *(const bf16x8*)(bp0 + kh * 16 + 8);
        fb1.h[0] = *(const bf16x8*)(bp1 + kh * 16);
        fb1.h[1] = *(const bf16x8*)(bp1 + kh * 16 + 8);

        acc0 = __builtin_amdgcn_wmma_f32_16x16x32_bf16(
            false, fa.v, false, fb0.v, (short)0, acc0, false, false);
        acc1 = __builtin_amdgcn_wmma_f32_16x16x32_bf16(
            false, fa.v, false, fb1.v, (short)0, acc1, false, false);
    }

    // C layout: VGPR j -> M = kh*8 + j ; N = lane&15
#pragma unroll
    for (int t = 0; t < 2; ++t) {
        const int col = colBase + wc * 32 + t * 16 + r;
        if (col < Nout) {
#pragma unroll
            for (int j = 0; j < 8; ++j) {
                int row = rowBase + wr * 16 + kh * 8 + j;
                float v = (t == 0) ? acc0[j] : acc1[j];
                if (row < M) C[(size_t)row * Nout + col] = v;
            }
        }
    }
}

// ---------------------------------------------------------------------------
// GAT attention pieces
// ---------------------------------------------------------------------------
__global__ void k_el_er(const float* __restrict__ feat, const float* __restrict__ al,
                        const float* __restrict__ ar, float* __restrict__ el,
                        float* __restrict__ er, int n, int heads) {
    long long total = (long long)n * heads;
    for (long long i = blockIdx.x * (long long)blockDim.x + threadIdx.x; i < total;
         i += (long long)gridDim.x * blockDim.x) {
        int node = (int)(i / heads), h = (int)(i % heads);
        const float* f = feat + ((size_t)node * heads + h) * 64;
        float sl = 0.f, sr = 0.f;
#pragma unroll
        for (int d = 0; d < 64; ++d) { sl += f[d] * al[h * 64 + d]; sr += f[d] * ar[h * 64 + d]; }
        el[i] = sl; er[i] = sr;
    }
}

__global__ void k_edge_max(const int* __restrict__ src, const int* __restrict__ dst,
                           const float* __restrict__ el, const float* __restrict__ er,
                           unsigned* __restrict__ m, long long total, int heads) {
    for (long long i = blockIdx.x * (long long)blockDim.x + threadIdx.x; i < total;
         i += (long long)gridDim.x * blockDim.x) {
        long long e = i / heads; int h = (int)(i % heads);
        float v = lrelu02(el[(size_t)src[e] * heads + h] + er[(size_t)dst[e] * heads + h]);
        atomicMax(m + (size_t)dst[e] * heads + h, f2ord(v));
    }
}

__global__ void k_edge_expsum(const int* __restrict__ src, const int* __restrict__ dst,
                              const float* __restrict__ el, const float* __restrict__ er,
                              const unsigned* __restrict__ m, float* __restrict__ denom,
                              float* __restrict__ ae, long long total, int heads) {
    for (long long i = blockIdx.x * (long long)blockDim.x + threadIdx.x; i < total;
         i += (long long)gridDim.x * blockDim.x) {
        long long e = i / heads; int h = (int)(i % heads);
        float v  = lrelu02(el[(size_t)src[e] * heads + h] + er[(size_t)dst[e] * heads + h]);
        float mm = ord2f(m[(size_t)dst[e] * heads + h]);
        float a  = __expf(v - mm);
        ae[i] = a;
        atomicAdd(denom + (size_t)dst[e] * heads + h, a);
    }
}

// one thread per (edge, head): alpha loaded once, feature row read as 16x b128
__global__ void k_aggregate(const int* __restrict__ src, const int* __restrict__ dst,
                            const float* __restrict__ ae, const float* __restrict__ denom,
                            const float* __restrict__ feat, float* __restrict__ out,
                            long long E, int heads) {
    long long total = E * heads;
    for (long long i = blockIdx.x * (long long)blockDim.x + threadIdx.x; i < total;
         i += (long long)gridDim.x * blockDim.x) {
        long long e = i / heads; int h = (int)(i % heads);
        int s = src[e], t = dst[e];
        float dn = denom[(size_t)t * heads + h];
        float alpha = ae[i] / (dn == 0.f ? 1.f : dn);
        const float4* f = (const float4*)(feat + ((size_t)s * heads + h) * 64);
        float* o = out + ((size_t)t * heads + h) * 64;
#pragma unroll
        for (int j = 0; j < 16; ++j) {
            float4 v = f[j];
            atomicAdd(o + 4 * j + 0, alpha * v.x);
            atomicAdd(o + 4 * j + 1, alpha * v.y);
            atomicAdd(o + 4 * j + 2, alpha * v.z);
            atomicAdd(o + 4 * j + 3, alpha * v.w);
        }
    }
}

__global__ void k_selu_bias(float* __restrict__ x, const float* __restrict__ bias,
                            long long count, int cols) {
    for (long long i = blockIdx.x * (long long)blockDim.x + threadIdx.x; i < count;
         i += (long long)gridDim.x * blockDim.x)
        x[i] = selu(x[i] + bias[i % cols]);
}

// ---------------------------------------------------------------------------
// Readout + MLP glue
// ---------------------------------------------------------------------------
__global__ void k_readout(const float* __restrict__ x, const float* __restrict__ sw,
                          const float* __restrict__ sb, const int* __restrict__ gid,
                          float* __restrict__ num, float* __restrict__ den, int n) {
    for (long long node = blockIdx.x * (long long)blockDim.x + threadIdx.x; node < n;
         node += (long long)gridDim.x * blockDim.x) {
        const float* f = x + node * 64;
        float s = sb[0];
#pragma unroll
        for (int d = 0; d < 64; ++d) s += f[d] * sw[d];
        float w = 1.f / (1.f + __expf(-s));
        int g = gid[node];
#pragma unroll
        for (int d = 0; d < 64; ++d) atomicAdd(num + (size_t)g * 64 + d, w * f[d]);
        atomicAdd(den + g, w);
    }
}

__global__ void k_concat(const float* __restrict__ num, const float* __restrict__ den,
                         const float* __restrict__ fg, bf16_t* __restrict__ z, int G) {
    long long total = (long long)G * 96;
    for (long long i = blockIdx.x * (long long)blockDim.x + threadIdx.x; i < total;
         i += (long long)gridDim.x * blockDim.x) {
        int g = (int)(i / 96), c = (int)(i % 96);
        float v;
        if (c < 64) { float dn = den[g]; v = num[(size_t)g * 64 + c] / (dn == 0.f ? 1.f : dn); }
        else if (c < 80) v = fg[(size_t)g * 16 + (c - 64)];
        else v = 0.f;
        z[i] = (bf16_t)v;
    }
}

__global__ void k_final(const float* __restrict__ h2, const float* __restrict__ w3,
                        const float* __restrict__ b3, float* __restrict__ out, int G) {
    for (long long g = blockIdx.x * (long long)blockDim.x + threadIdx.x; g < G;
         g += (long long)gridDim.x * blockDim.x) {
        float s = b3[0];
#pragma unroll
        for (int d = 0; d < 64; ++d) s += h2[g * 64 + d] * w3[d];
        out[g] = s;
    }
}

// ---------------------------------------------------------------------------
// Host orchestration
// ---------------------------------------------------------------------------
static inline int nblk(long long n, int tb) {
    long long b = (n + tb - 1) / tb;
    if (b > (1 << 20)) b = (1 << 20);
    if (b < 1) b = 1;
    return (int)b;
}

extern "C" void kernel_launch(void* const* d_in, const int* in_sizes, int n_in,
                              void* d_out, int out_size, void* d_ws, size_t ws_size,
                              hipStream_t stream) {
    const float* feats_node  = (const float*)d_in[0];
    const float* feats_graph = (const float*)d_in[1];
    const int*   src   = (const int*)d_in[2];
    const int*   dst   = (const int*)d_in[3];
    const int*   gid   = (const int*)d_in[4];
    const float* fc1_w = (const float*)d_in[5];
    const float* al1   = (const float*)d_in[6];
    const float* ar1   = (const float*)d_in[7];
    const float* bias1 = (const float*)d_in[8];
    const float* fc2_w = (const float*)d_in[9];
    const float* al2   = (const float*)d_in[10];
    const float* ar2   = (const float*)d_in[11];
    const float* bias2 = (const float*)d_in[12];
    const float* fc3_w = (const float*)d_in[13];
    const float* al3   = (const float*)d_in[14];
    const float* ar3   = (const float*)d_in[15];
    const float* bias3 = (const float*)d_in[16];
    const float* score_w = (const float*)d_in[17];
    const float* score_b = (const float*)d_in[18];
    const float* mlp_w1  = (const float*)d_in[19];
    const float* mlp_b1  = (const float*)d_in[20];
    const float* mlp_w2  = (const float*)d_in[21];
    const float* mlp_b2  = (const float*)d_in[22];
    const float* mlp_w3  = (const float*)d_in[23];
    const float* mlp_b3  = (const float*)d_in[24];

    const int N = in_sizes[0] / 128;
    const int G = in_sizes[1] / 16;
    const int E = in_sizes[2];
    const int TB = 256;

    // workspace carve-out
    size_t off = 0;
    auto alloc = [&](size_t bytes) -> void* {
        void* p = (char*)d_ws + off;
        off = (off + bytes + 255) & ~(size_t)255;
        return p;
    };
    float*    featP = (float*)   alloc((size_t)N * 256 * 4);
    float*    agg   = (float*)   alloc((size_t)N * 256 * 4);
    bf16_t*   xb    = (bf16_t*)  alloc((size_t)N * 256 * 2);
    bf16_t*   wb    = (bf16_t*)  alloc((size_t)256 * 256 * 2);
    float*    el    = (float*)   alloc((size_t)N * 4 * 4);
    float*    er    = (float*)   alloc((size_t)N * 4 * 4);
    unsigned* mmax  = (unsigned*)alloc((size_t)N * 4 * 4);
    float*    denom = (float*)   alloc((size_t)N * 4 * 4);
    float*    ae    = (float*)   alloc((size_t)E * 4 * 4);
    float*    gnum  = (float*)   alloc((size_t)G * 64 * 4);
    float*    gden  = (float*)   alloc((size_t)G * 4);
    bf16_t*   zb    = (bf16_t*)  alloc((size_t)G * 96 * 2);
    float*    h1    = (float*)   alloc((size_t)G * 128 * 4);
    bf16_t*   h1b   = (bf16_t*)  alloc((size_t)G * 128 * 2);
    float*    h2    = (float*)   alloc((size_t)G * 64 * 4);

    auto gat_edges = [&](const float* feat, const float* al, const float* ar,
                         const float* bias, int heads) {
        int hd = heads * 64;
        k_el_er<<<nblk((long long)N * heads, TB), TB, 0, stream>>>(feat, al, ar, el, er, N, heads);
        k_fill_u32<<<nblk((long long)N * heads, TB), TB, 0, stream>>>(mmax, 0u, (long long)N * heads);
        k_fill_u32<<<nblk((long long)N * heads, TB), TB, 0, stream>>>((unsigned*)denom, 0u, (long long)N * heads);
        k_fill_u32<<<nblk((long long)N * hd, TB), TB, 0, stream>>>((unsigned*)agg, 0u, (long long)N * hd);
        k_edge_max<<<nblk((long long)E * heads, TB), TB, 0, stream>>>(src, dst, el, er, mmax, (long long)E * heads, heads);
        k_edge_expsum<<<nblk((long long)E * heads, TB), TB, 0, stream>>>(src, dst, el, er, mmax, denom, ae, (long long)E * heads, heads);
        k_aggregate<<<nblk((long long)E * heads, TB), TB, 0, stream>>>(src, dst, ae, denom, feat, agg, E, heads);
        k_selu_bias<<<nblk((long long)N * hd, TB), TB, 0, stream>>>(agg, bias, (long long)N * hd, hd);
    };
    auto gemm = [&](const bf16_t* A, const bf16_t* Bw, float* C, int M, int Nout, int Kpad) {
        int nTN = Nout / 64; if (nTN < 1) nTN = 1;
        dim3 grid(((M + 63) / 64) * nTN);
        k_gemm_bf16<<<grid, 256, 0, stream>>>(A, Bw, C, M, Nout, Kpad, nTN);
    };

    // ---- Layer 1: [N,128] -> [N,256] ----
    k_f32_to_bf16_pad<<<nblk((long long)N * 128, TB), TB, 0, stream>>>(feats_node, xb, N, 128, 128);
    k_f32_to_bf16_pad<<<nblk(256 * 128, TB), TB, 0, stream>>>(fc1_w, wb, 256, 128, 128);
    gemm(xb, wb, featP, N, 256, 128);
    gat_edges(featP, al1, ar1, bias1, 4);

    // ---- Layer 2: [N,256] -> [N,256] ----
    k_f32_to_bf16_pad<<<nblk((long long)N * 256, TB), TB, 0, stream>>>(agg, xb, N, 256, 256);
    k_f32_to_bf16_pad<<<nblk(256 * 256, TB), TB, 0, stream>>>(fc2_w, wb, 256, 256, 256);
    gemm(xb, wb, featP, N, 256, 256);
    gat_edges(featP, al2, ar2, bias2, 4);

    // ---- Layer 3: [N,256] -> [N,64], 1 head ----
    k_f32_to_bf16_pad<<<nblk((long long)N * 256, TB), TB, 0, stream>>>(agg, xb, N, 256, 256);
    k_f32_to_bf16_pad<<<nblk(64 * 256, TB), TB, 0, stream>>>(fc3_w, wb, 64, 256, 256);
    gemm(xb, wb, featP, N, 64, 256);
    gat_edges(featP, al3, ar3, bias3, 1);

    // ---- Weighted-average readout -> [G,64], concat extra -> bf16 [G,96] ----
    k_fill_u32<<<nblk((long long)G * 64, TB), TB, 0, stream>>>((unsigned*)gnum, 0u, (long long)G * 64);
    k_fill_u32<<<nblk(G, TB), TB, 0, stream>>>((unsigned*)gden, 0u, G);
    k_readout<<<nblk(N, TB), TB, 0, stream>>>(agg, score_w, score_b, gid, gnum, gden, N);
    k_concat<<<nblk((long long)G * 96, TB), TB, 0, stream>>>(gnum, gden, feats_graph, zb, G);

    // ---- MLP: [G,80(->96)] -> 128 -> 64 -> 1 ----
    k_f32_to_bf16_pad<<<nblk(128 * 96, TB), TB, 0, stream>>>(mlp_w1, wb, 128, 80, 96);
    gemm(zb, wb, h1, G, 128, 96);
    k_selu_bias<<<nblk((long long)G * 128, TB), TB, 0, stream>>>(h1, mlp_b1, (long long)G * 128, 128);
    k_f32_to_bf16_pad<<<nblk((long long)G * 128, TB), TB, 0, stream>>>(h1, h1b, G, 128, 128);
    k_f32_to_bf16_pad<<<nblk(64 * 128, TB), TB, 0, stream>>>(mlp_w2, wb, 64, 128, 128);
    gemm(h1b, wb, h2, G, 64, 128);
    k_selu_bias<<<nblk((long long)G * 64, TB), TB, 0, stream>>>(h2, mlp_b2, (long long)G * 64, 64);
    k_final<<<nblk(G, TB), TB, 0, stream>>>(h2, mlp_w3, mlp_b3, (float*)d_out, G);
}
